// NeRFW3D_79370995630430
// MI455X (gfx1250) — matrix operand order repaired and checked by reading, not compiled, not run on hardware
//
#include <hip/hip_runtime.h>
#include <hip/hip_bf16.h>

// ---------------------------------------------------------------------------
// NeRF-W forward, CDNA5 (gfx1250, wave32) WMMA implementation.
//   hT = W x actT per 16-row tile; activations stay in VGPRs across all layers.
//   All repacked f16 weights (321.5 KB) are staged global->LDS once per block
//   with a single Tensor-Data-Mover op (TENSOR_LOAD_TO_LDS + s_wait_tensorcnt),
//   A-fragments then stream from LDS (ds_load_b128 pairs).
// ---------------------------------------------------------------------------

typedef __attribute__((ext_vector_type(16))) _Float16     v16h;
typedef __attribute__((ext_vector_type(8)))  _Float16     v8h;
typedef __attribute__((ext_vector_type(8)))  float        v8f;
typedef __attribute__((ext_vector_type(8)))  unsigned int v8u;
typedef __attribute__((ext_vector_type(4)))  unsigned int v4u;
typedef __attribute__((ext_vector_type(8)))  int          v8i;
typedef __attribute__((ext_vector_type(4)))  int          v4i;

#define NPTS 262144

// f16-element offsets of repacked weights inside d_ws (mirrored 1:1 into LDS)
#define OFF_W0   0        // 128 x 64   (63 -> pad 64)
#define OFF_W1   8192     // 128 x 128
#define OFF_W2   24576
#define OFF_W3   40960
#define OFF_W4   57344    // 128 x 192  (191 -> pad 192)
#define OFF_W5   81920
#define OFF_W6   98304
#define OFF_W7   114688
#define OFF_FT   131072   // feat 128 x 128
#define OFF_R1   147456   // rgb1 64 x 192 (feat|d_enc(pad32)|app)
#define OFF_R2   159744   // rgb2 16 x 64  (3 rows valid)
#define WS_F16_TOTAL 160768
#define SMEM_BYTES (WS_F16_TOTAL * 2)      // 321536 B <= 320 KB WGP LDS
#define OFF_R2B_BYTES (WS_F16_TOTAL * 2)   // 16 floats, padded rgb2 bias (in ws)

// ------------------------------ helpers ------------------------------------

__device__ inline unsigned pk2(float a, float b) {
    union { _Float16 h[2]; unsigned u; } r;
    r.h[0] = (_Float16)a; r.h[1] = (_Float16)b;
    return r.u;
}

// Convert a pair of D tiles (out^T, f32) into one B fragment (32K x 16N, f16)
// for the next layer.  D: lanes<16 hold ch r, lanes>=16 hold ch r+8.
// B: lanes<16 need K0..15 (dword v = K 2v,2v+1), lanes>=16 need K16..31.
__device__ inline v16h dpair_to_b(const v8f d0, const v8f d1, bool hi, bool do_relu) {
    unsigned p0[4], p1[4];
#pragma unroll
    for (int v = 0; v < 4; ++v) {
        float a0 = d0[2 * v], a1 = d0[2 * v + 1];
        float c0 = d1[2 * v], c1 = d1[2 * v + 1];
        if (do_relu) {
            a0 = fmaxf(a0, 0.f); a1 = fmaxf(a1, 0.f);
            c0 = fmaxf(c0, 0.f); c1 = fmaxf(c1, 0.f);
        }
        p0[v] = pk2(a0, a1);
        p1[v] = pk2(c0, c1);
    }
    v8u u;
#pragma unroll
    for (int v = 0; v < 4; ++v) {
        unsigned q0 = __shfl_xor(p0[v], 16, 32);   // partner half-wave
        unsigned q1 = __shfl_xor(p1[v], 16, 32);
        u[v]     = hi ? q1     : p0[v];
        u[4 + v] = hi ? p1[v]  : q0;
    }
    return __builtin_bit_cast(v16h, u);
}

template <int NC>
__device__ inline void conv_to_b(const v8f* acc, v16h* B, bool hi, bool do_relu) {
#pragma unroll
    for (int c = 0; c < NC; ++c)
        B[c] = dpair_to_b(acc[2 * c], acc[2 * c + 1], hi, do_relu);
}

// One dense layer: out^T(16NT x 16rows, f32) = W(A) x act^T(B) + bias.
// W is f16 in LDS, row-major [out][k], k-stride kp (multiple of 32).
template <int NT, int KC>
__device__ inline void mlp_layer(const _Float16* __restrict__ W, int kp,
                                 const float* __restrict__ bias,
                                 const v16h* __restrict__ B, v8f* __restrict__ acc,
                                 int lane) {
    const int hi8  = (lane >> 4) * 8;   // 0 or 8
    const int mloc = lane & 15;
#pragma unroll
    for (int t = 0; t < NT; ++t) {
        const float* bp = bias + t * 16 + hi8;
        float4 bA = *(const float4*)bp;
        float4 bB = *(const float4*)(bp + 4);
        v8f c = { bA.x, bA.y, bA.z, bA.w, bB.x, bB.y, bB.z, bB.w };
#pragma unroll
        for (int kc = 0; kc < KC; ++kc) {
            // A fragment (16x32 f16): lane reads K run of 16 at 32kc+2*hi8.
            const _Float16* wp = W + (size_t)(t * 16 + mloc) * kp + kc * 32 + 2 * hi8;
            v8h a0 = *(const v8h*)wp;          // ds_load_b128
            v8h a1 = *(const v8h*)(wp + 8);    // ds_load_b128
            v16h A = __builtin_shufflevector(a0, a1, 0, 1, 2, 3, 4, 5, 6, 7,
                                             8, 9, 10, 11, 12, 13, 14, 15);
            c = __builtin_amdgcn_wmma_f32_16x16x32_f16(
                    false, A, false, B[kc], (short)0, c, false, false);
        }
        acc[t] = c;
    }
}

// NeRF positional-encoding feature j of [x, sin/cos(2^l * x_c)...]
__device__ inline float encf(int j, int nfeat, float a0, float a1, float a2) {
    if (j >= nfeat) return 0.f;
    if (j < 3) return (j == 0) ? a0 : ((j == 1) ? a1 : a2);
    int jm = j - 3;
    int l = jm / 6;
    int rem = jm - 6 * l;
    int s = rem / 3;              // 0=sin, 1=cos
    int c = rem - 3 * s;
    float xv = (c == 0) ? a0 : ((c == 1) ? a1 : a2);
    float arg = xv * (float)(1 << l);
    return s ? __cosf(arg) : __sinf(arg);
}

// Build encoding directly in B-fragment layout for K chunk starting at kbase.
__device__ inline v16h enc_frag(int kbase, int nfeat, float a0, float a1, float a2) {
    v8u u;
#pragma unroll
    for (int v = 0; v < 8; ++v)
        u[v] = pk2(encf(kbase + 2 * v,     nfeat, a0, a1, a2),
                   encf(kbase + 2 * v + 1, nfeat, a0, a1, a2));
    return __builtin_bit_cast(v16h, u);
}

// ------------------------------ weight prep --------------------------------

__global__ void pack_w(const float* __restrict__ src, _Float16* __restrict__ dst,
                       int out_dim, int in_dim, int in_pad) {
    int i = blockIdx.x * blockDim.x + threadIdx.x;
    int total = out_dim * in_pad;
    if (i >= total) return;
    int m = i / in_pad, k = i - m * in_pad;
    float v = (k < in_dim) ? src[m * in_dim + k] : 0.f;
    dst[i] = (_Float16)v;
}

// rgb1: 64 x 187 -> 64 x 192, columns remapped to [feat 0..127 | d_enc 128..154,
// pad 155..159 | app 160..191]
__global__ void pack_rgb1(const float* __restrict__ src, _Float16* __restrict__ dst) {
    int i = blockIdx.x * blockDim.x + threadIdx.x;
    if (i >= 64 * 192) return;
    int m = i / 192, k = i - m * 192;
    float v = 0.f;
    if (k < 155)       v = src[m * 187 + k];
    else if (k >= 160) v = src[m * 187 + 155 + (k - 160)];
    dst[i] = (_Float16)v;
}

// rgb2: 3 x 64 -> 16 x 64, rows 3..15 zero
__global__ void pack_rgb2w(const float* __restrict__ src, _Float16* __restrict__ dst) {
    int i = blockIdx.x * blockDim.x + threadIdx.x;
    if (i >= 16 * 64) return;
    int m = i >> 6, k = i & 63;
    dst[i] = (m < 3) ? (_Float16)src[m * 64 + k] : (_Float16)0.f;
}

__global__ void pack_rgb2b(const float* __restrict__ src, float* __restrict__ dst) {
    int i = threadIdx.x;
    if (i < 16) dst[i] = (i < 3) ? src[i] : 0.f;
}

// ------------------------------ main kernel --------------------------------

struct KParams {
    const float* x; const float* d; const int* idx;
    const _Float16* w;                 // repacked weights in d_ws (global)
    const float* b[8];
    const float* sigW; const float* sigB;
    const float* featB; const float* rgb1B; const float* rgb2B;
    const float* app;
    float* out;
};

__global__ __launch_bounds__(256)
void nerf_fwd(KParams p) {
    extern __shared__ char smem_raw[];                 // SMEM_BYTES of weights
    _Float16* lw = (_Float16*)smem_raw;

    const int lane = threadIdx.x & 31;
    const int wave = threadIdx.x >> 5;
    const bool hi  = lane >= 16;
    const int  hi8 = hi ? 8 : 0;
    const int  n   = (blockIdx.x * 8 + wave) * 16 + (lane & 15);  // this lane's row

    // ---- TDM: stage the whole weight image global -> LDS (one DMA, wave 0)
    if (wave == 0) {
        const unsigned total_qw = SMEM_BYTES / 8;      // 40192 8-byte elements
        unsigned long long ga = (unsigned long long)(uintptr_t)p.w;
        unsigned lds_base = (unsigned)(uintptr_t)smem_raw;
        // D# group 0: count=1 | lds_addr | global_addr | type=2
        v4u g0 = { 1u,
                   lds_base,
                   (unsigned)(ga & 0xffffffffu),
                   (unsigned)((ga >> 32) & 0x01ffffffu) | 0x80000000u };
        // D# group 1: data_size=8B (code 3); 1-D tensor/tile of total_qw elems
        v8i g1;
        g1[0] = 0x00030000;                               // data_size=3
        g1[1] = (int)((total_qw & 0xffffu) << 16);        // tensor_dim0[15:0]
        g1[2] = (int)(((total_qw >> 16) & 0xffffu)        // tensor_dim0[31:16]
                      | (1u << 16));                      // tensor_dim1 = 1
        g1[3] = (int)((total_qw & 0xffffu) << 16);        // tile_dim0
        g1[4] = 0;                                        // tile_dim1/2 unused
        g1[5] = (int)total_qw;                            // tensor_dim0_stride
        g1[6] = 0;
        g1[7] = 0;
        v4i gz = { 0, 0, 0, 0 };
#if defined(__clang_major__) && (__clang_major__ >= 23)
        v8i g4 = { 0, 0, 0, 0, 0, 0, 0, 0 };
        __builtin_amdgcn_tensor_load_to_lds(g0, g1, gz, gz, g4, 0);
#else
        __builtin_amdgcn_tensor_load_to_lds(g0, g1, gz, gz, 0);
#endif
        __builtin_amdgcn_s_wait_tensorcnt(0);
    }
    __syncthreads();                                   // weights visible to all waves

    // ---- positional encoding of x, directly as B fragments (2 chunks, K pad 64)
    float x0 = p.x[3 * n], x1 = p.x[3 * n + 1], x2 = p.x[3 * n + 2];
    v16h xB[2];
    xB[0] = enc_frag(0  + (hi ? 16 : 0), 63, x0, x1, x2);
    xB[1] = enc_frag(32 + (hi ? 16 : 0), 63, x0, x1, x2);

    v8f  acc[8];
    v16h B[6];

    // ---- pts MLP, 8 layers, skip-concat before layer 4
    mlp_layer<8, 2>(lw + OFF_W0, 64,  p.b[0], xB, acc, lane);
    conv_to_b<4>(acc, B, hi, true);
    mlp_layer<8, 4>(lw + OFF_W1, 128, p.b[1], B, acc, lane);
    conv_to_b<4>(acc, B, hi, true);
    mlp_layer<8, 4>(lw + OFF_W2, 128, p.b[2], B, acc, lane);
    conv_to_b<4>(acc, B, hi, true);
    mlp_layer<8, 4>(lw + OFF_W3, 128, p.b[3], B, acc, lane);
    conv_to_b<4>(acc, B, hi, true);

    B[4] = xB[0]; B[5] = xB[1];                        // concat([h, x_enc])
    mlp_layer<8, 6>(lw + OFF_W4, 192, p.b[4], B, acc, lane);
    conv_to_b<4>(acc, B, hi, true);

    mlp_layer<8, 4>(lw + OFF_W5, 128, p.b[5], B, acc, lane);
    conv_to_b<4>(acc, B, hi, true);
    mlp_layer<8, 4>(lw + OFF_W6, 128, p.b[6], B, acc, lane);
    conv_to_b<4>(acc, B, hi, true);
    mlp_layer<8, 4>(lw + OFF_W7, 128, p.b[7], B, acc, lane);
    conv_to_b<4>(acc, B, hi, true);                    // h (relu) as B frags

    // ---- sigma = relu(h . sigW + b): each lane holds 64 of 128 channels
    float sp = 0.f;
#pragma unroll
    for (int t = 0; t < 8; ++t) {
        const float* swp = p.sigW + t * 16 + hi8;
        float4 s0 = *(const float4*)swp;
        float4 s1 = *(const float4*)(swp + 4);
        v8f h = acc[t];
        sp += fmaxf(h[0], 0.f) * s0.x + fmaxf(h[1], 0.f) * s0.y
            + fmaxf(h[2], 0.f) * s0.z + fmaxf(h[3], 0.f) * s0.w
            + fmaxf(h[4], 0.f) * s1.x + fmaxf(h[5], 0.f) * s1.y
            + fmaxf(h[6], 0.f) * s1.z + fmaxf(h[7], 0.f) * s1.w;
    }
    sp += __shfl_xor(sp, 16, 32);
    float sig = fmaxf(sp + p.sigB[0], 0.f);

    // ---- feat (no activation) -> B frags
    mlp_layer<8, 4>(lw + OFF_FT, 128, p.featB, B, acc, lane);
    conv_to_b<4>(acc, B, hi, false);

    // ---- d_enc chunk (27 feats, pad 32) and appearance chunk (32)
    float d0 = p.d[3 * n], d1 = p.d[3 * n + 1], d2 = p.d[3 * n + 2];
    B[4] = enc_frag(hi ? 16 : 0, 27, d0, d1, d2);
    {
        const float* ar = p.app + (size_t)p.idx[n] * 32 + (hi ? 16 : 0);
        v8u u;
#pragma unroll
        for (int v = 0; v < 8; ++v) u[v] = pk2(ar[2 * v], ar[2 * v + 1]);
        B[5] = __builtin_bit_cast(v16h, u);
    }

    // ---- rgb head
    mlp_layer<4, 6>(lw + OFF_R1, 192, p.rgb1B, B, acc, lane);
    conv_to_b<2>(acc, B, hi, true);
    mlp_layer<1, 2>(lw + OFF_R2, 64,  p.rgb2B, B, acc, lane);

    float r0 = 1.f / (1.f + __expf(-acc[0][0]));
    float r1 = 1.f / (1.f + __expf(-acc[0][1]));
    float r2 = 1.f / (1.f + __expf(-acc[0][2]));

    if (lane < 16) {                                   // lanes<16 hold M=0..7 (rgb in 0..2)
        p.out[3 * n]     = r0;
        p.out[3 * n + 1] = r1;
        p.out[3 * n + 2] = r2;
        p.out[3 * NPTS + n] = sig;                     // sigma after rgb block
    }
}

// ------------------------------ launcher -----------------------------------

extern "C" void kernel_launch(void* const* d_in, const int* in_sizes, int n_in,
                              void* d_out, int out_size, void* d_ws, size_t ws_size,
                              hipStream_t stream) {
    (void)in_sizes; (void)n_in; (void)out_size; (void)ws_size;
    // d_in order: 0 x, 1 d, 2 img_idx, 3..18 (W_i,b_i)*8, 19 sigma_W, 20 sigma_b,
    //             21 feat_W, 22 feat_b, 23 rgb1_W, 24 rgb1_b, 25 rgb2_W, 26 rgb2_b,
    //             27 app_table
    _Float16* ws = (_Float16*)d_ws;
    float* rgb2b = (float*)((char*)d_ws + OFF_R2B_BYTES);
    auto F = [&](int i) { return (const float*)d_in[i]; };

    pack_w<<<(128 * 64  + 255) / 256, 256, 0, stream>>>(F(3),  ws + OFF_W0, 128, 63, 64);
    pack_w<<<(128 * 128 + 255) / 256, 256, 0, stream>>>(F(5),  ws + OFF_W1, 128, 128, 128);
    pack_w<<<(128 * 128 + 255) / 256, 256, 0, stream>>>(F(7),  ws + OFF_W2, 128, 128, 128);
    pack_w<<<(128 * 128 + 255) / 256, 256, 0, stream>>>(F(9),  ws + OFF_W3, 128, 128, 128);
    pack_w<<<(128 * 192 + 255) / 256, 256, 0, stream>>>(F(11), ws + OFF_W4, 128, 191, 192);
    pack_w<<<(128 * 128 + 255) / 256, 256, 0, stream>>>(F(13), ws + OFF_W5, 128, 128, 128);
    pack_w<<<(128 * 128 + 255) / 256, 256, 0, stream>>>(F(15), ws + OFF_W6, 128, 128, 128);
    pack_w<<<(128 * 128 + 255) / 256, 256, 0, stream>>>(F(17), ws + OFF_W7, 128, 128, 128);
    pack_w<<<(128 * 128 + 255) / 256, 256, 0, stream>>>(F(21), ws + OFF_FT, 128, 128, 128);
    pack_rgb1 <<<(64 * 192 + 255) / 256, 256, 0, stream>>>(F(23), ws + OFF_R1);
    pack_rgb2w<<<(16 * 64  + 255) / 256, 256, 0, stream>>>(F(25), ws + OFF_R2);
    pack_rgb2b<<<1, 16, 0, stream>>>(F(26), rgb2b);

    KParams p;
    p.x = F(0); p.d = F(1); p.idx = (const int*)d_in[2];
    p.w = ws;
    p.b[0] = F(4);  p.b[1] = F(6);  p.b[2] = F(8);  p.b[3] = F(10);
    p.b[4] = F(12); p.b[5] = F(14); p.b[6] = F(16); p.b[7] = F(18);
    p.sigW = F(19); p.sigB = F(20);
    p.featB = F(22); p.rgb1B = F(24); p.rgb2B = rgb2b;
    p.app = F(27);
    p.out = (float*)d_out;

    // allow 321.5 KB dynamic LDS (WGP has 320 KB for the single resident group)
    hipFuncSetAttribute(reinterpret_cast<const void*>(nerf_fwd),
                        hipFuncAttributeMaxDynamicSharedMemorySize, SMEM_BYTES);

    nerf_fwd<<<NPTS / 128, 256, SMEM_BYTES, stream>>>(p);
}
